// CustomCrossAttention_26980984553755
// MI455X (gfx1250) — matrix operand order repaired and checked by reading, not compiled, not run on hardware
//
#include <hip/hip_runtime.h>
#include <hip/hip_bf16.h>

// MI455X (gfx1250) cross-attention, f16 WMMA everywhere, fp32 accum.
// Memory-bound on the 1 GiB attn_weights store; K/V are L2-resident so the
// two-pass (recompute) softmax adds no HBM traffic.

typedef _Float16 v8h  __attribute__((ext_vector_type(8)));
typedef _Float16 v16h __attribute__((ext_vector_type(16)));
typedef float    v8f  __attribute__((ext_vector_type(8)));

#define CAT16(lo, hi) __builtin_shufflevector(lo, hi, 0,1,2,3,4,5,6,7,8,9,10,11,12,13,14,15)

static __device__ __forceinline__ v8f wmma_f16(v16h a, v16h b, v8f c) {
  // v_wmma_f32_16x16x32_f16  (8-arg form: neg_a, A, neg_b, B, c_mod, C, reuse_a, reuse_b)
  return __builtin_amdgcn_wmma_f32_16x16x32_f16(false, a, false, b, (short)0, c, false, false);
}

// A-operand (M=16 x K=32, f16) per ISA 7.12.2: lane l -> row m=l%16, g=l/16,
// cols {8g..8g+7} then {16+8g..16+8g+7}.  `a` points at (row0, k0) of a
// row-major [*, ld] f16 matrix; ld multiple of 8, base 16B-aligned.
static __device__ __forceinline__ v16h load_a16(const _Float16* a, int ld, int lane) {
  const int m = lane & 15, g = lane >> 4;
  const _Float16* p = a + (size_t)m * ld;
  v8h lo = *(const v8h*)(p + 8 * g);
  v8h hi = *(const v8h*)(p + 16 + 8 * g);
  return CAT16(lo, hi);
}

// Same A layout but converting from fp32 source on the fly.
static __device__ __forceinline__ v16h load_a_f32(const float* a, int ld, int k0, int lane) {
  const int m = lane & 15, g = lane >> 4;
  const float* p = a + (size_t)m * ld + k0;
  v16h r;
#pragma unroll
  for (int i = 0; i < 8; ++i) r[i] = (_Float16)p[8 * g + i];
#pragma unroll
  for (int i = 0; i < 8; ++i) r[8 + i] = (_Float16)p[16 + 8 * g + i];
  return r;
}

// B-operand (K=32 x N=16, f16): lane l -> col n=l%16, g=l/16, rows K={8g..8g+7}
// then {16+8g..16+8g+7} (same k order as the A operand).  `bt` is a
// "B-transposed" row-major [N][K] view pointing at (k0, n0); lane reads 8 f16
// at bt + n*ldn + 8g and 8 f16 at bt + n*ldn + 16 + 8g.
static __device__ __forceinline__ v16h load_b16(const _Float16* bt, int ldn, int lane) {
  const int n = lane & 15, g = lane >> 4;
  const _Float16* p = bt + (size_t)n * ldn + 8 * g;
  v8h lo = *(const v8h*)(p);
  v8h hi = *(const v8h*)(p + 16);
  return CAT16(lo, hi);
}

// ---------------- weight transpose / f16 convert ----------------
__global__ void tr_wq_kernel(const float* __restrict__ W, _Float16* __restrict__ WT) {
  int i = blockIdx.x * blockDim.x + threadIdx.x;          // WT: (1024 e, 64 d)
  if (i < 64 * 1024) { int e = i >> 6, d = i & 63; WT[i] = (_Float16)W[d * 1024 + e]; }
}
__global__ void tr_wc_kernel(const float* __restrict__ W, _Float16* __restrict__ WT) {
  int i = blockIdx.x * blockDim.x + threadIdx.x;          // WT: (2048 e, 128 d)
  if (i < 128 * 2048) { int e = i >> 7, d = i & 127; WT[i] = (_Float16)W[d * 2048 + e]; }
}
__global__ void tr_wp_kernel(const float* __restrict__ W, _Float16* __restrict__ WT) {
  int i = blockIdx.x * blockDim.x + threadIdx.x;          // WT: (64 d, 1024 e)
  if (i < 64 * 1024) { int d = i >> 10, e = i & 1023; WT[i] = (_Float16)W[e * 64 + d]; }
}

// ---------------- Q projection: (8192x64) @ (64x1024) + bq, *scale ----------------
__global__ void qproj_kernel(const float* __restrict__ hs, const _Float16* __restrict__ wqT,
                             const float* __restrict__ bq, _Float16* __restrict__ q16) {
  const int lane = threadIdx.x & 31, wave = threadIdx.x >> 5;
  const int tile = blockIdx.x * 4 + wave;                 // 512 mtiles x 64 ntiles
  const int mt = tile >> 6, nt = tile & 63;
  const int n = lane & 15, g = lane >> 4;

  const float*    arow = hs  + (size_t)mt * 16 * 64;
  const _Float16* bt   = wqT + (size_t)nt * 16 * 64;
  v8f c = {};
  c = wmma_f16(load_a_f32(arow, 64, 0,  lane), load_b16(bt,      64, lane), c);
  c = wmma_f16(load_a_f32(arow, 64, 32, lane), load_b16(bt + 32, 64, lane), c);

  const int e = nt * 16 + n, h = e >> 6, d = e & 63;
  const float bias = bq[e];
#pragma unroll
  for (int r = 0; r < 8; ++r) {
    int row = mt * 16 + r + 8 * g;                        // row = b*2048 + lq
    int b = row >> 11, lq = row & 2047;
    q16[((size_t)(b * 16 + h) * 2048 + lq) * 64 + d] = (_Float16)((c[r] + bias) * 0.125f);
  }
}

// ---------------- KV projection: (8192x128) @ (128x2048) + bc ----------------
__global__ void kvproj_kernel(const float* __restrict__ enc, const _Float16* __restrict__ wcT,
                              const float* __restrict__ bc, _Float16* __restrict__ k16,
                              _Float16* __restrict__ vT16) {
  const int lane = threadIdx.x & 31, wave = threadIdx.x >> 5;
  const int tile = blockIdx.x * 4 + wave;                 // 512 mtiles x 128 ntiles
  const int mt = tile >> 7, nt = tile & 127;
  const int n = lane & 15, g = lane >> 4;

  const float*    arow = enc + (size_t)mt * 16 * 128;
  const _Float16* bt   = wcT + (size_t)nt * 16 * 128;
  v8f c = {};
#pragma unroll
  for (int ch = 0; ch < 4; ++ch)
    c = wmma_f16(load_a_f32(arow, 128, ch * 32, lane), load_b16(bt + ch * 32, 128, lane), c);

  const int e = nt * 16 + n;
  const float bias = bc[e];
  if (nt < 64) {                                          // K half -> (bh, lk, d)
    const int h = e >> 6, d = e & 63;
#pragma unroll
    for (int r = 0; r < 8; ++r) {
      int row = mt * 16 + r + 8 * g;
      int b = row >> 11, lk = row & 2047;
      k16[((size_t)(b * 16 + h) * 2048 + lk) * 64 + d] = (_Float16)(c[r] + bias);
    }
  } else {                                                // V half -> transposed (bh, d, lk)
    const int ev = e - 1024, h = ev >> 6, d = ev & 63;
    int row0 = mt * 16 + 8 * g;                           // 8 consecutive lk per lane
    int b = row0 >> 11, lk0 = row0 & 2047;
    v8h pk;
#pragma unroll
    for (int r = 0; r < 8; ++r) pk[r] = (_Float16)(c[r] + bias);
    *(v8h*)(vT16 + ((size_t)(b * 16 + h) * 64 + d) * 2048 + lk0) = pk;
  }
}

// ---------------- attention: per-wave 16-row q strip, two-pass softmax ----------------
__global__ void attn_kernel(const _Float16* __restrict__ q16, const _Float16* __restrict__ k16,
                            const _Float16* __restrict__ vT16, const float* __restrict__ mask,
                            float* __restrict__ attnW, _Float16* __restrict__ ctx16) {
  __shared__ __align__(16) _Float16 pstage[4][16 * 40];   // per-wave P staging (C->A transpose)
  const int lane = threadIdx.x & 31, wave = threadIdx.x >> 5;
  const int strip = blockIdx.x * 4 + wave;                // B*H*(Lq/16) = 8192 strips
  const int bh = strip >> 7, qt = strip & 127, b = bh >> 4;
  const int n = lane & 15, g = lane >> 4;

  const _Float16* qb = q16  + ((size_t)bh * 2048 + qt * 16) * 64;
  const _Float16* kb = k16  + (size_t)bh * 2048 * 64;
  const _Float16* vb = vT16 + (size_t)bh * 64 * 2048;
  const float*    mb = mask + (size_t)b * 2048;

  const v16h aq0 = load_a16(qb, 64, lane);                // Q stays resident in VGPRs
  const v16h aq1 = load_a16(qb + 32, 64, lane);

  float lm[8], ls[8];
#pragma unroll
  for (int r = 0; r < 8; ++r) { lm[r] = -3.0e38f; ls[r] = 0.0f; }

  // ---- pass 1: online per-lane (max, sum) over this lane's key column ----
  for (int t = 0; t < 128; ++t) {
    const _Float16* kt = kb + (size_t)(t * 16) * 64;
    v8f c = {};
    c = wmma_f16(aq0, load_b16(kt,      64, lane), c);
    c = wmma_f16(aq1, load_b16(kt + 32, 64, lane), c);
    float mk = mb[t * 16 + n];
#pragma unroll
    for (int r = 0; r < 8; ++r) {
      float x = c[r] + mk;
      float nm = fmaxf(lm[r], x);
      ls[r] = ls[r] * __expf(lm[r] - nm) + __expf(x - nm);
      lm[r] = nm;
    }
  }
  // merge across the 16 lanes sharing each row (xor<16 stays within the half-wave)
#pragma unroll
  for (int r = 0; r < 8; ++r) {
#pragma unroll
    for (int s = 0; s < 4; ++s) {
      int off = 1 << s;
      float om = __shfl_xor(lm[r], off);
      float os = __shfl_xor(ls[r], off);
      float nm = fmaxf(lm[r], om);
      ls[r] = ls[r] * __expf(lm[r] - nm) + os * __expf(om - nm);
      lm[r] = nm;
    }
    ls[r] = 1.0f / ls[r];                                 // 1/rowsum
  }

  // ---- pass 2: recompute scores, emit weights, accumulate ctx = P @ V ----
  v8f acc0 = {}, acc1 = {}, acc2 = {}, acc3 = {};
  _Float16* sw = &pstage[wave][0];
  for (int ch = 0; ch < 64; ++ch) {                       // 32 keys per chunk
#pragma unroll
    for (int half = 0; half < 2; ++half) {
      int t = ch * 2 + half;
      const _Float16* kt = kb + (size_t)(t * 16) * 64;
      v8f c = {};
      c = wmma_f16(aq0, load_b16(kt,      64, lane), c);
      c = wmma_f16(aq1, load_b16(kt + 32, 64, lane), c);
      float mk = mb[t * 16 + n];
#pragma unroll
      for (int r = 0; r < 8; ++r) {
        float w = __expf(c[r] + mk - lm[r]) * ls[r];
        int row = r + 8 * g;
        attnW[((size_t)bh * 2048 + qt * 16 + row) * 2048 + t * 16 + n] = w;
        sw[row * 40 + half * 16 + n] = (_Float16)w;       // stage for C->A transpose
      }
    }
    v16h ap = load_a16(sw, 40, lane);                     // P as A-operand (K=32)
    const _Float16* vch = vb + ch * 32;
    acc0 = wmma_f16(ap, load_b16(vch,             2048, lane), acc0);
    acc1 = wmma_f16(ap, load_b16(vch + 16 * 2048, 2048, lane), acc1);
    acc2 = wmma_f16(ap, load_b16(vch + 32 * 2048, 2048, lane), acc2);
    acc3 = wmma_f16(ap, load_b16(vch + 48 * 2048, 2048, lane), acc3);
  }

  // store ctx in merged-head layout (B, Lq, 1024) as f16 for the output GEMM
  const int h = bh & 15;
#pragma unroll
  for (int r = 0; r < 8; ++r) {
    int lq = qt * 16 + r + 8 * g;
    size_t ro = ((size_t)b * 2048 + lq) * 1024 + h * 64;
    ctx16[ro + n]      = (_Float16)acc0[r];
    ctx16[ro + 16 + n] = (_Float16)acc1[r];
    ctx16[ro + 32 + n] = (_Float16)acc2[r];
    ctx16[ro + 48 + n] = (_Float16)acc3[r];
  }
}

// ---------------- output projection: (8192x1024) @ (1024x64) + bp ----------------
__global__ void oproj_kernel(const _Float16* __restrict__ ctx16, const _Float16* __restrict__ wpT,
                             const float* __restrict__ bp, float* __restrict__ out) {
  const int lane = threadIdx.x & 31, wave = threadIdx.x >> 5;
  const int tile = blockIdx.x * 4 + wave;                 // 512 mtiles x 4 ntiles
  const int mt = tile >> 2, nt = tile & 3;
  const int n = lane & 15, g = lane >> 4;

  const _Float16* arow = ctx16 + (size_t)mt * 16 * 1024;
  const _Float16* bt   = wpT   + (size_t)nt * 16 * 1024;
  v8f c = {};
  for (int ch = 0; ch < 32; ++ch)
    c = wmma_f16(load_a16(arow + ch * 32, 1024, lane), load_b16(bt + ch * 32, 1024, lane), c);

  const float bias = bp[nt * 16 + n];
#pragma unroll
  for (int r = 0; r < 8; ++r) {
    int row = mt * 16 + r + 8 * g;
    out[(size_t)row * 64 + nt * 16 + n] = c[r] + bias;
  }
}

extern "C" void kernel_launch(void* const* d_in, const int* in_sizes, int n_in,
                              void* d_out, int out_size, void* d_ws, size_t ws_size,
                              hipStream_t stream) {
  (void)in_sizes; (void)n_in; (void)out_size; (void)ws_size;
  const float* hs   = (const float*)d_in[0];   // (4,2048,64)
  const float* enc  = (const float*)d_in[1];   // (4,2048,128)
  const float* mask = (const float*)d_in[2];   // (4,1,1,2048)
  const float* Wq   = (const float*)d_in[3];   // (64,1024)
  const float* bq   = (const float*)d_in[4];
  const float* Wc   = (const float*)d_in[5];   // (128,2048)
  const float* bc   = (const float*)d_in[6];
  const float* Wp   = (const float*)d_in[7];   // (1024,64)
  const float* bp   = (const float*)d_in[8];

  float* out   = (float*)d_out;                       // attn_output: 4*2048*64
  float* attnW = out + (size_t)4 * 2048 * 64;         // attn_weights: 4*16*2048*2048

  char* ws = (char*)d_ws;                             // ~65 MB total
  const size_t SZ = (size_t)8192 * 1024 * sizeof(_Float16);   // 16 MiB each
  _Float16* q16  = (_Float16*)(ws);
  _Float16* k16  = (_Float16*)(ws + SZ);
  _Float16* vT16 = (_Float16*)(ws + 2 * SZ);
  _Float16* c16  = (_Float16*)(ws + 3 * SZ);
  _Float16* wqT  = (_Float16*)(ws + 4 * SZ);
  _Float16* wcT  = (_Float16*)(ws + 4 * SZ + (size_t)64 * 1024 * 2);
  _Float16* wpT  = (_Float16*)(ws + 4 * SZ + ((size_t)64 * 1024 + 128 * 2048) * 2);

  tr_wq_kernel<<<(64 * 1024) / 256, 256, 0, stream>>>(Wq, wqT);
  tr_wc_kernel<<<(128 * 2048) / 256, 256, 0, stream>>>(Wc, wcT);
  tr_wp_kernel<<<(64 * 1024) / 256, 256, 0, stream>>>(Wp, wpT);
  qproj_kernel <<<(512 * 64) / 4, 128, 0, stream>>>(hs, wqT, bq, q16);
  kvproj_kernel<<<(512 * 128) / 4, 128, 0, stream>>>(enc, wcT, bc, k16, vT16);
  attn_kernel  <<<8192 / 4, 128, 0, stream>>>(q16, k16, vT16, mask, attnW, c16);
  oproj_kernel <<<(512 * 4) / 4, 128, 0, stream>>>(c16, wpT, bp, out);
}
